// Merge_18545668784379
// MI455X (gfx1250) — compile-verified, matrix operand order checked by reading, not run on hardware
//
#include <hip/hip_runtime.h>
#include <hip/hip_bf16.h>

// ---------------------------------------------------------------------------
// CDNA5 (gfx1250) implementation of the table-structure network forward pass.
// Convs are LDS-staged implicit GEMMs on v_wmma_f32_16x16x32_f16 (wave32).
// Block = 8x16 output pixels x 16 output channels; input tile (+/-3 halo) and
// weights staged once in LDS as f16; hot loop is vector ds_loads + WMMA.
// ---------------------------------------------------------------------------

typedef __attribute__((ext_vector_type(16))) _Float16 v16h;
typedef __attribute__((ext_vector_type(8)))  _Float16 v8h;
typedef __attribute__((ext_vector_type(8)))  float    v8f;
typedef __attribute__((ext_vector_type(4)))  int      v4i;

#define ZPITCH 262144  // per-plane pitch of zone-sum buffer (>= max zr*nzc+zc)

// Staged tile geometry: 16x8 outputs, halo 3 each side, row stride 24.
#define TLW   24                 // LDS row stride (22 used cols, padded)
#define TLH   14                 // 8 + 2*3 rows
#define TPLN  336                // TLW*TLH per input channel
#define CINMX 37
#define KPMX  896                // max padded K (18*49 -> 896)

// ---------------------------------------------------------------------------
// Generic implicit-GEMM convolution (same-padded: OH==H, OW==W).
//   out[b, outCoff+n, y, x] = act( bias[n] + sum_k in[...] * w[n, k] )
// K = Cin*ksz*ksz in OIHW flattening order.
// ---------------------------------------------------------------------------
__global__ __launch_bounds__(128)
void conv_wmma(const float* __restrict__ in, const float* __restrict__ wgt,
               const float* __restrict__ bias, float* __restrict__ out,
               int Bn, int inCtot, int inCoff, int Cin, int H, int W,
               int Cout, int outCtot, int outCoff,
               int ksz, int dil, int pad, int relu, int tilesY)
{
    __shared__ __align__(16) _Float16 alds[CINMX * TPLN + TPLN]; // + zero region
    __shared__ __align__(16) _Float16 wlds[16 * KPMX];
    __shared__ __align__(16) int      tab[KPMX];

    const int HW     = H * W;
    const int kk     = ksz * ksz;
    const int Ktot   = Cin * kk;
    const int chunks = (Ktot + 31) >> 5;
    const int KPAD   = chunks << 5;
    const int tid    = threadIdx.x;

    const int tx    = blockIdx.x;                 // 16-wide x tile
    const int b     = blockIdx.y / tilesY;        // batch
    const int ty    = blockIdx.y - b * tilesY;    // 8-tall y tile
    const int x0    = tx * 16, y0 = ty * 8;
    const int nBase = blockIdx.z * 16;

    __builtin_prefetch(wgt + (long)nBase * Ktot, 0, 3);  // global_prefetch_b8

    // ---- stage input tile (f16, zero-filled OOB halo) ----
    const int cbase = (b * inCtot + inCoff) * HW;
    for (int ci = 0; ci < Cin; ++ci) {
        const int pb = cbase + ci * HW;
        for (int pos = tid; pos < TLH * 22; pos += 128) {
            const int ly = pos / 22, lx = pos - ly * 22;
            const int iy = y0 - 3 + ly, ix = x0 - 3 + lx;
            float v = 0.f;
            if (iy >= 0 && iy < H && ix >= 0 && ix < W) v = in[pb + iy * W + ix];
            alds[ci * TPLN + ly * TLW + lx] = (_Float16)v;
        }
    }
    for (int i = tid; i < TPLN; i += 128)            // dedicated zero region
        alds[CINMX * TPLN + i] = (_Float16)0.f;

    // ---- stage 16 weight rows (f16), zero-padded ----
    for (int n = 0; n < 16; ++n) {
        const int  nc = nBase + n;
        const long wb = (long)nc * Ktot;
        for (int k = tid; k < KPAD; k += 128)
            wlds[n * KPAD + k] = (nc < Cout && k < Ktot) ? (_Float16)wgt[wb + k]
                                                         : (_Float16)0.f;
    }

    // ---- K -> LDS-offset table (only divisions in the kernel; cold) ----
    for (int k = tid; k < KPAD; k += 128) {
        int e = CINMX * TPLN;                         // pad -> zero region
        if (k < Ktot) {
            const int ci = k / kk;
            const int r  = k - ci * kk;
            const int kh = r / ksz;
            const int kw = r - kh * ksz;
            e = ci * TPLN + (kh * dil - pad + 3) * TLW + (kw * dil - pad + 3);
        }
        tab[k] = e;
    }
    __syncthreads();

    // ---- WMMA main loop: wave = 2 row-tiles (rows 2w, 2w+1) x 16 channels ----
    const int lane = tid & 31;
    const int wave = tid >> 5;
    const int l15  = lane & 15;
    const int hiK  = (lane >> 4) << 3;     // lanes 16-31 own K+8 / K+24
    const int hiM  = (lane >> 4) << 3;     // C: lanes 16-31 -> M+8
    const int ry0  = wave * 2, ry1 = ry0 + 1;
    const _Float16* a0p = alds + (ry0 * TLW + l15); // lane pixel base pointers
    const _Float16* a1p = alds + (ry1 * TLW + l15);
    const _Float16* wrow = &wlds[l15 * KPAD];

    v8f acc0 = {0.f, 0.f, 0.f, 0.f, 0.f, 0.f, 0.f, 0.f};
    v8f acc1 = {0.f, 0.f, 0.f, 0.f, 0.f, 0.f, 0.f, 0.f};

    for (int c = 0; c < chunks; ++c) {
        const int k0 = (c << 5) + hiK;
        // K-table for this chunk: two contiguous 8-int runs -> 4x ds_load_b128.
        const v4i t0 = *reinterpret_cast<const v4i*>(&tab[k0]);
        const v4i t1 = *reinterpret_cast<const v4i*>(&tab[k0 + 4]);
        const v4i t2 = *reinterpret_cast<const v4i*>(&tab[k0 + 16]);
        const v4i t3 = *reinterpret_cast<const v4i*>(&tab[k0 + 20]);
        int kof[16];
        kof[0]  = t0.x; kof[1]  = t0.y; kof[2]  = t0.z; kof[3]  = t0.w;
        kof[4]  = t1.x; kof[5]  = t1.y; kof[6]  = t1.z; kof[7]  = t1.w;
        kof[8]  = t2.x; kof[9]  = t2.y; kof[10] = t2.z; kof[11] = t2.w;
        kof[12] = t3.x; kof[13] = t3.y; kof[14] = t3.z; kof[15] = t3.w;

        v16h av0, av1;
#pragma unroll
        for (int h = 0; h < 16; ++h) {
            // ISA 16-bit A layout: halves 0-7 -> K 0..7, 8-15 -> K 16..23
            av0[h] = a0p[kof[h]];
            av1[h] = a1p[kof[h]];
        }
        // B fragment: two contiguous 8-half LDS vector loads (ds_load_b128)
        const v8h lo = *reinterpret_cast<const v8h*>(wrow + k0);
        const v8h hi = *reinterpret_cast<const v8h*>(wrow + k0 + 16);
        const v16h bv = __builtin_shufflevector(lo, hi,
                            0, 1, 2, 3, 4, 5, 6, 7, 8, 9, 10, 11, 12, 13, 14, 15);
        acc0 = __builtin_amdgcn_wmma_f32_16x16x32_f16(
                   false, av0, false, bv, (short)0, acc0, false, false);
        acc1 = __builtin_amdgcn_wmma_f32_16x16x32_f16(
                   false, av1, false, bv, (short)0, acc1, false, false);
    }

    // ---- epilogue: lane = channel, VGPR j = pixel x (j+8 for lanes 16-31) ----
    const int  nCol = nBase + l15;
    const bool nval = (nCol < Cout);
    const float bs  = nval ? bias[nCol] : 0.f;
    if (nval) {
#pragma unroll
        for (int t = 0; t < 2; ++t) {
            const int y = y0 + (t ? ry1 : ry0);
            if (y < H) {
                const v8f a = t ? acc1 : acc0;
                const int ob = ((b * outCtot + outCoff + nCol) * H + y) * W;
#pragma unroll
                for (int j = 0; j < 8; ++j) {
                    const int x = x0 + hiM + j;
                    if (x < W) {
                        float v = a[j] + bs;
                        if (relu) v = fmaxf(v, 0.f);
                        out[ob + x] = v;
                    }
                }
            }
        }
    }
}

// ---------------------------------------------------------------------------
// AvgPool2d(kernel=2, stride=1, VALID): [BC,H,W] -> [BC,H-1,W-1], *0.25
// ---------------------------------------------------------------------------
__global__ void avgpool2_k(const float* __restrict__ in, float* __restrict__ out,
                           int BC, int H, int W)
{
    const int OH = H - 1, OW = W - 1;
    const long total = (long)BC * OH * OW;
    for (long i = (long)blockIdx.x * blockDim.x + threadIdx.x; i < total;
         i += (long)gridDim.x * blockDim.x) {
        const int p = (int)(i / ((long)OH * OW));
        const int r = (int)(i - (long)p * OH * OW);
        const int y = r / OW, x = r - y * OW;
        const float* ip = in + ((long)p * H + y) * W + x;
        out[i] = 0.25f * (ip[0] + ip[1] + ip[W] + ip[W + 1]);
    }
}

__global__ void fill0_k(float* __restrict__ p, long n)
{
    for (long i = (long)blockIdx.x * blockDim.x + threadIdx.x; i < n;
         i += (long)gridDim.x * blockDim.x)
        p[i] = 0.f;
}

// ---------------------------------------------------------------------------
// Zone segmentation from grid_struc [HP,WP] (int32).
// seg layout (ints): [0..511] row_seg  [512..1023] col_seg
//                    [1024..1535] rcount  [1536..2047] ccount  [2048] nzc
// ---------------------------------------------------------------------------
__global__ __launch_bounds__(512)
void seg_build_k(const int* __restrict__ g, int* __restrict__ seg, int HP, int WP)
{
    __shared__ int rsep[512], csep[512];
    const int t = threadIdx.x;
    int all;
    if (t < HP) {
        all = 1;
        for (int x = 0; x < WP; ++x) all &= (g[t * WP + x] != 0);
        rsep[t] = (t > 0) ? all : 0;
    } else rsep[t] = 0;
    if (t < WP) {
        all = 1;
        for (int y = 0; y < HP; ++y) all &= (g[y * WP + t] != 0);
        csep[t] = (t > 0) ? all : 0;
    } else csep[t] = 0;
    __syncthreads();
    if (t == 0) {
        int s = 0;
        for (int i = 0; i < 512; ++i) { if (i < HP) s += rsep[i]; seg[i] = (i < HP) ? s : 0; }
        s = 0;
        for (int i = 0; i < 512; ++i) { if (i < WP) s += csep[i]; seg[512 + i] = (i < WP) ? s : 0; }
        for (int i = 0; i < 512; ++i) { seg[1024 + i] = 0; seg[1536 + i] = 0; }
        for (int i = 0; i < HP; ++i) seg[1024 + seg[i]]++;
        for (int i = 0; i < WP; ++i) seg[1536 + seg[512 + i]]++;
        seg[2048] = seg[512 + WP - 1] + 1;   // number of column zones
    }
}

// Accumulate per-zone sums: in is contiguous [planes, H, W].
__global__ void pool_sum_k(const float* __restrict__ in, float* __restrict__ zsum,
                           const int* __restrict__ seg, int planes, int H, int W)
{
    const int nzc = seg[2048];
    const long total = (long)planes * H * W;
    for (long i = (long)blockIdx.x * blockDim.x + threadIdx.x; i < total;
         i += (long)gridDim.x * blockDim.x) {
        const int p = (int)(i / ((long)H * W));
        const int r = (int)(i - (long)p * H * W);
        const int y = r / W, x = r - y * W;
        const int z = seg[y] * nzc + seg[512 + x];
        atomicAdd(&zsum[(long)p * ZPITCH + z], in[i]);
    }
}

// Broadcast zone means (optionally sigmoid) into a channel slice of `out`;
// optional second contiguous destination (for bp2 outputs, C==1).
__global__ void pool_bcast_k(const float* __restrict__ zsum, const int* __restrict__ seg,
                             float* __restrict__ out, float* __restrict__ out2,
                             int Bn, int C, int outCtot, int outCoff,
                             int H, int W, int sig)
{
    const int nzc = seg[2048];
    const int* rc = seg + 1024;
    const int* cc = seg + 1536;
    const long total = (long)Bn * C * H * W;
    for (long i = (long)blockIdx.x * blockDim.x + threadIdx.x; i < total;
         i += (long)gridDim.x * blockDim.x) {
        const int p = (int)(i / ((long)H * W));      // plane = b*C + c
        const int r = (int)(i - (long)p * H * W);
        const int y = r / W, x = r - y * W;
        const int zr = seg[y], zc = seg[512 + x];
        const float cnt = (float)(rc[zr] * cc[zc]);
        float v = zsum[(long)p * ZPITCH + zr * nzc + zc] / cnt;
        if (sig) v = 1.f / (1.f + __expf(-v));
        const int b = p / C, c = p - b * C;
        out[(((long)b * outCtot + outCoff + c) * H + y) * W + x] = v;
        if (out2) out2[i] = v;
    }
}

// ---------------------------------------------------------------------------
// Host orchestration
// ---------------------------------------------------------------------------
extern "C" void kernel_launch(void* const* d_in, const int* in_sizes, int n_in,
                              void* d_out, int out_size, void* d_ws, size_t ws_size,
                              hipStream_t stream)
{
    (void)in_sizes; (void)n_in; (void)out_size; (void)ws_size;

    const int Bn = 2;
    const int HP = 510, WP = 510;
    const long HWo = (long)HP * WP;          // 260100
    const long SLOT = (long)Bn * HWo;        // per-output-slot floats

    // ---- workspace layout (floats) ----
    const long SZB = 9437184;                // 2*18*512*512
    float* R0 = (float*)d_ws;                // ping
    float* R1 = R0 + SZB;                    // pong
    float* R2 = R1 + SZB;                    // feat [2,18,510,510]
    float* R3 = R2 + SZB;                    // concat buf A [2,37,510,510]
    float* R4 = R3 + 19248000;               // concat buf B [2,37,510,510]
    float* ZS = R4 + 19248000;               // zone sums (planes x ZPITCH)
    int*   SEG = (int*)(ZS + SZB);           // segmentation ints (2049)

    const float* input = (const float*)d_in[0];
    const int*   grid  = (const int*)d_in[1];
    float*       outp  = (float*)d_out;

    // Param flattening assumption: {"input","grid_struc","params"}; params dict
    // in sorted-key (jax pytree) order: down, left, right, sfcn, up.
    // Branch = 3 blocks x {br1,br2,c1,c2,c3} x {b,w}.  sfcn = 4 x {b,w}.
    const int PB_DOWN = 2, PB_LEFT = 32, PB_RIGHT = 62, PB_SFCN = 92, PB_UP = 100;
    auto P = [&](int i) -> const float* { return (const float*)d_in[i]; };

    auto conv = [&](const float* in, int inCtot, int inCoff, int Cin, int H, int W,
                    const float* w, const float* b, int Cout,
                    float* out, int outCtot, int outCoff,
                    int ksz, int dil, int pad, int relu) {
        const int tilesX = (W + 15) / 16;
        const int tilesY = (H + 7) / 8;
        dim3 gridd(tilesX, tilesY * Bn, (Cout + 15) / 16);
        conv_wmma<<<gridd, 128, 0, stream>>>(in, w, b, out, Bn, inCtot, inCoff, Cin,
                                             H, W, Cout, outCtot, outCoff,
                                             ksz, dil, pad, relu, tilesY);
    };

    auto pool = [&](const float* src, int C, float* dst, int outCtot, int outCoff,
                    int sig, float* out2) {
        const int planes = Bn * C;
        fill0_k<<<1024, 256, 0, stream>>>(ZS, (long)planes * ZPITCH);
        pool_sum_k<<<2048, 256, 0, stream>>>(src, ZS, SEG, planes, HP, WP);
        pool_bcast_k<<<2048, 256, 0, stream>>>(ZS, SEG, dst, out2, Bn, C,
                                               outCtot, outCoff, HP, WP, sig);
    };

    // ---- zone segmentation ----
    seg_build_k<<<1, 512, 0, stream>>>(grid, SEG, HP, WP);

    // ---- SFCN: 4x conv7x7(pad3)+ReLU, pools after layers 1 and 3 ----
    conv(input, 8, 0, 8, 512, 512, P(PB_SFCN + 1), P(PB_SFCN + 0), 18, R0, 18, 0, 7, 1, 3, 1);
    conv(R0, 18, 0, 18, 512, 512, P(PB_SFCN + 3), P(PB_SFCN + 2), 18, R1, 18, 0, 7, 1, 3, 1);
    avgpool2_k<<<2048, 256, 0, stream>>>(R1, R0, Bn * 18, 512, 512);          // -> 511
    conv(R0, 18, 0, 18, 511, 511, P(PB_SFCN + 5), P(PB_SFCN + 4), 18, R1, 18, 0, 7, 1, 3, 1);
    conv(R1, 18, 0, 18, 511, 511, P(PB_SFCN + 7), P(PB_SFCN + 6), 18, R0, 18, 0, 7, 1, 3, 1);
    avgpool2_k<<<2048, 256, 0, stream>>>(R0, R2, Bn * 18, 511, 511);          // feat @510

    // ---- one directional branch ----
    auto run_branch = [&](int pb, int slot) {
        auto BK = [&](int blk, int off) { return pb + blk * 10 + off; };
        // block 1: in = feat R2 (18ch) -> concat R3 (37ch)
        conv(R2, 18, 0, 18, HP, WP, P(BK(0, 5)), P(BK(0, 4)), 6, R3, 37, 18, 3, 1, 1, 0);
        conv(R2, 18, 0, 18, HP, WP, P(BK(0, 7)), P(BK(0, 6)), 6, R3, 37, 24, 3, 2, 2, 0);
        conv(R2, 18, 0, 18, HP, WP, P(BK(0, 9)), P(BK(0, 8)), 6, R3, 37, 30, 3, 3, 3, 0);
        conv(R3, 37, 18, 18, HP, WP, P(BK(0, 1)), P(BK(0, 0)), 18, R0, 18, 0, 1, 1, 0, 0);
        pool(R0, 18, R3, 37, 0, 0, nullptr);                                   // b1
        conv(R3, 37, 18, 18, HP, WP, P(BK(0, 3)), P(BK(0, 2)), 1, R1, 1, 0, 1, 1, 0, 0);
        pool(R1, 1, R3, 37, 36, 1, nullptr);                                   // b2
        // block 2: in = R3 (37ch) -> concat R4; b2 also -> output slot (bp2)
        conv(R3, 37, 0, 37, HP, WP, P(BK(1, 5)), P(BK(1, 4)), 6, R4, 37, 18, 3, 1, 1, 0);
        conv(R3, 37, 0, 37, HP, WP, P(BK(1, 7)), P(BK(1, 6)), 6, R4, 37, 24, 3, 2, 2, 0);
        conv(R3, 37, 0, 37, HP, WP, P(BK(1, 9)), P(BK(1, 8)), 6, R4, 37, 30, 3, 3, 3, 0);
        conv(R4, 37, 18, 18, HP, WP, P(BK(1, 1)), P(BK(1, 0)), 18, R0, 18, 0, 1, 1, 0, 0);
        pool(R0, 18, R4, 37, 0, 0, nullptr);
        conv(R4, 37, 18, 18, HP, WP, P(BK(1, 3)), P(BK(1, 2)), 1, R1, 1, 0, 1, 1, 0, 0);
        pool(R1, 1, R4, 37, 36, 1, outp + (long)slot * SLOT);                  // bp2
        // block 3: in = R4; only b2 needed (bp3). br1 / concat skipped (dead).
        conv(R4, 37, 0, 37, HP, WP, P(BK(2, 5)), P(BK(2, 4)), 6, R0, 18, 0, 3, 1, 1, 0);
        conv(R4, 37, 0, 37, HP, WP, P(BK(2, 7)), P(BK(2, 6)), 6, R0, 18, 6, 3, 2, 2, 0);
        conv(R4, 37, 0, 37, HP, WP, P(BK(2, 9)), P(BK(2, 8)), 6, R0, 18, 12, 3, 3, 3, 0);
        conv(R0, 18, 0, 18, HP, WP, P(BK(2, 3)), P(BK(2, 2)), 1, R1, 1, 0, 1, 1, 0, 0);
        pool(R1, 1, outp + (long)(slot + 1) * SLOT, 1, 0, 1, nullptr);         // bp3
    };

    run_branch(PB_UP,    0);
    run_branch(PB_DOWN,  2);
    run_branch(PB_LEFT,  4);
    run_branch(PB_RIGHT, 6);
}
